// MixtureOfExperts_2542620639799
// MI455X (gfx1250) — matrix operand order repaired and checked by reading, not compiled, not run on hardware
//
#include <hip/hip_runtime.h>
#include <hip/hip_bf16.h>

typedef __attribute__((ext_vector_type(16))) __bf16 v16bf;
typedef __attribute__((ext_vector_type(8)))  float  v8f;

#define NTOK   8192
#define CDIM   768
#define FDIM   3072
#define NEXP   8
#define FCHUNK 768
#define NCHUNK 4            // FDIM / FCHUNK
#define XSTR   776          // padded LDS row stride (ushorts)
#define TPW    6            // 48 col-tiles of 16 / 8 waves
#define CAP    (2 * NTOK)   // per-expert bucket capacity (worst case)
#define MAXTILES (CAP / 16 + NEXP - 1)   // 1031
#define GBLKS  (NTOK / 256)              // 32 gate blocks

struct __align__(16) U4 { unsigned int x, y, z, w; };
union AFrag { v16bf v; unsigned int u[8]; };
union BFrag { v16bf v; U4 q[2]; };

__device__ __forceinline__ unsigned short f2bf(float f) {
  unsigned int u = __float_as_uint(f);
  unsigned int r = u + 0x7FFFu + ((u >> 16) & 1u);   // round-to-nearest-even
  return (unsigned short)(r >> 16);
}

// ---------------------------------------------------------------------------
// 1) f32 -> bf16 convert + weight transposes; zero bucket counters
//    xb  [N][C]     = bf16(x)
//    w1t [E][F][C]  = bf16(w1[e][c][f])   (K contiguous for stage-1 B)
//    w2t [E][C][F]  = bf16(w2[e][f][c])   (K contiguous for stage-2 B)
// ---------------------------------------------------------------------------
__global__ __launch_bounds__(256) void moe_convert_kernel(
    const float* __restrict__ x, const float* __restrict__ w1,
    const float* __restrict__ w2, unsigned short* __restrict__ xb,
    unsigned short* __restrict__ w1t, unsigned short* __restrict__ w2t,
    int* __restrict__ cnt) {
  int i = blockIdx.x * blockDim.x + threadIdx.x;
  int stride = gridDim.x * blockDim.x;
  if (i < NEXP) cnt[i] = 0;
  const int nx = NTOK * CDIM;
  const int nw = NEXP * CDIM * FDIM;
  const int total = nx + 2 * nw;
  for (int o = i; o < total; o += stride) {
    if (o < nx) {
      xb[o] = f2bf(x[o]);
    } else if (o < nx + nw) {
      int r0 = o - nx;
      int e = r0 / (CDIM * FDIM);
      int r = r0 % (CDIM * FDIM);
      int f = r / CDIM, c = r % CDIM;
      w1t[r0] = f2bf(w1[e * CDIM * FDIM + c * FDIM + f]);
    } else {
      int r0 = o - nx - nw;
      int e = r0 / (FDIM * CDIM);
      int r = r0 % (FDIM * CDIM);
      int c = r / FDIM, f = r % FDIM;
      w2t[r0] = f2bf(w2[e * FDIM * CDIM + f * CDIM + c]);
    }
  }
}

// ---------------------------------------------------------------------------
// 2) Gating: top-2 softmax -> per-expert buckets; deterministic aux sums
//    bucket entry: (tok | k<<16), weight
// ---------------------------------------------------------------------------
__global__ __launch_bounds__(256) void moe_gate_kernel(
    const float* __restrict__ x, const float* __restrict__ gw,
    int* __restrict__ cnt, int* __restrict__ btok, float* __restrict__ bw,
    float* __restrict__ blocksums) {
  __shared__ float red[256];
  int tid = threadIdx.x;
  int n = blockIdx.x * 256 + tid;

  float acc[NEXP];
#pragma unroll
  for (int e = 0; e < NEXP; ++e) acc[e] = 0.f;
  const float* xr = x + (long long)n * CDIM;
  for (int c = 0; c < CDIM; ++c) {
    float xv = xr[c];
#pragma unroll
    for (int e = 0; e < NEXP; ++e) acc[e] = fmaf(xv, gw[c * NEXP + e], acc[e]);
  }
  // top-2 (descending, earliest index on ties, like lax.top_k)
  int i0 = 0; float l0 = acc[0];
#pragma unroll
  for (int e = 1; e < NEXP; ++e) if (acc[e] > l0) { l0 = acc[e]; i0 = e; }
  int i1 = -1; float l1 = -3.4e38f;
#pragma unroll
  for (int e = 0; e < NEXP; ++e)
    if (e != i0 && acc[e] > l1) { l1 = acc[e]; i1 = e; }
  float ew  = __expf(l1 - l0);
  float inv = 1.f / (1.f + ew);

  // scatter into expert buckets (slot order is non-deterministic, but each
  // row's GEMM result is independent of tile placement -> values are stable)
  int s0 = atomicAdd(&cnt[i0], 1);
  btok[i0 * CAP + s0] = n;                 // k = 0 in bits [16+]
  bw[i0 * CAP + s0]   = inv;
  int s1 = atomicAdd(&cnt[i1], 1);
  btok[i1 * CAP + s1] = n | (1 << 16);     // k = 1
  bw[i1 * CAP + s1]   = ew * inv;

  // full-8 fp32 softmax, deterministic per-block tree reduction
  float s = 0.f, p[NEXP];
#pragma unroll
  for (int e = 0; e < NEXP; ++e) { p[e] = __expf(acc[e] - l0); s += p[e]; }
  float is = 1.f / s;
  for (int e = 0; e < NEXP; ++e) {
    red[tid] = p[e] * is;
    __syncthreads();
    for (int off = 128; off > 0; off >>= 1) {
      if (tid < off) red[tid] += red[tid + off];
      __syncthreads();
    }
    if (tid == 0) blocksums[blockIdx.x * NEXP + e] = red[0];
    __syncthreads();
  }
}

// ---------------------------------------------------------------------------
// 3) Prefix-sum of padded tile counts (1 thread)
// ---------------------------------------------------------------------------
__global__ void moe_tileoff_kernel(const int* __restrict__ cnt,
                                   int* __restrict__ tileoff) {
  int off = 0;
  for (int e = 0; e < NEXP; ++e) {
    tileoff[e] = off;
    off += (cnt[e] + 15) / 16;
  }
  tileoff[NEXP] = off;
}

// ---------------------------------------------------------------------------
// 4) Expert MLP: one (expert, 16-token tile) per block, WMMA bf16 -> f32
//    Writes weighted rows into yp[k][token][col]  (disjoint, no atomics)
// ---------------------------------------------------------------------------
__global__ __launch_bounds__(256) void moe_mlp_kernel(
    const unsigned short* __restrict__ xb, const unsigned short* __restrict__ w1t,
    const unsigned short* __restrict__ w2t, const int* __restrict__ btok,
    const float* __restrict__ bw, const int* __restrict__ cnt,
    const int* __restrict__ tileoff, float* __restrict__ yp) {
  __shared__ unsigned short lds_x[16 * XSTR];   // 24.8 KB
  __shared__ unsigned short lds_h[16 * XSTR];   // 24.8 KB
  __shared__ int   soff[NEXP + 1];
  __shared__ int   scnt[NEXP];
  __shared__ int   tokL[16];
  __shared__ int   kL[16];
  __shared__ float wL[16];

  const int tid  = threadIdx.x;
  const int wave = tid >> 5;
  const int lane = tid & 31;
  const int m16  = lane & 15;                   // A row / B,D column
  const int s    = lane >> 4;                   // lane half
  const int bid  = blockIdx.x;

  if (tid < NEXP + 1) soff[tid] = tileoff[tid];
  if (tid < NEXP)     scnt[tid] = cnt[tid];
  __syncthreads();
  if (bid >= soff[NEXP]) return;                // uniform per block

  int e = 0;
  while (bid >= soff[e + 1]) ++e;               // uniform (shared data)
  const int t = bid - soff[e];
  const int cntE = scnt[e];

  if (tid < 16) {
    int st = t * 16 + tid;
    if (st < cntE) {
      int pk = btok[e * CAP + st];
      tokL[tid] = pk & 0xFFFF;
      kL[tid]   = pk >> 16;
      wL[tid]   = bw[e * CAP + st];
    } else {
      tokL[tid] = -1; kL[tid] = 0; wL[tid] = 0.f;
    }
  }
  __syncthreads();

  // gather the 16 bf16 x rows into LDS (padded rows read row 0, weight 0)
#pragma unroll
  for (int k = 0; k < 6; ++k) {
    int idx = tid + k * 256;                    // 1536 x 8-ushort chunks
    int eo = idx * 8;
    int r = eo / CDIM, c = eo % CDIM;
    int tk = tokL[r];
    int src = (tk < 0) ? 0 : tk;
    *(U4*)&lds_x[r * XSTR + c] = *(const U4*)&xb[src * CDIM + c];
  }
  __syncthreads();

  const unsigned short* w1e = w1t + e * FDIM * CDIM;
  const unsigned short* w2e = w2t + e * CDIM * FDIM;

  v8f zero = {};
  v8f y_acc[TPW];
#pragma unroll
  for (int tt = 0; tt < TPW; ++tt) y_acc[tt] = zero;

  for (int fc = 0; fc < NCHUNK; ++fc) {
    const int f0 = fc * FCHUNK;
    // ---- stage 1: h_chunk = gelu(x @ w1[:, f0:f0+768]) ----
#pragma unroll
    for (int tt = 0; tt < TPW; ++tt) {
      const int j = (wave * TPW + tt) * 16;     // col within chunk
      const unsigned short* bptr = w1e + (f0 + j + m16) * CDIM + s * 8;
      v8f acc = zero;
      for (int kk = 0; kk < CDIM; kk += 32) {
        AFrag a;
#pragma unroll
        for (int v = 0; v < 8; ++v) {
          int kb = ((v >> 2) << 4) + (s << 3) + ((v & 3) << 1) + kk;
          a.u[v] = *(const unsigned int*)&lds_x[m16 * XSTR + kb];
        }
        BFrag b;
        b.q[0] = *(const U4*)&bptr[kk];
        b.q[1] = *(const U4*)&bptr[kk + 16];
        acc = __builtin_amdgcn_wmma_f32_16x16x32_bf16(
            false, a.v, false, b.v, (short)0, acc, false, false);
      }
#pragma unroll
      for (int v = 0; v < 8; ++v) {
        float xv = acc[v];
        float g = 0.5f * xv * (1.f + erff(xv * 0.70710678118654752f));
        lds_h[(v + 8 * s) * XSTR + j + m16] = f2bf(g);
      }
    }
    __syncthreads();
    // ---- stage 2: y += h_chunk @ w2[f0:f0+768, :] ----
#pragma unroll
    for (int tt = 0; tt < TPW; ++tt) {
      const int j = (wave * TPW + tt) * 16;     // output column
      const unsigned short* bptr = w2e + (j + m16) * FDIM + f0 + s * 8;
      v8f acc = y_acc[tt];
      for (int kk = 0; kk < FCHUNK; kk += 32) {
        AFrag a;
#pragma unroll
        for (int v = 0; v < 8; ++v) {
          int kb = ((v >> 2) << 4) + (s << 3) + ((v & 3) << 1) + kk;
          a.u[v] = *(const unsigned int*)&lds_h[m16 * XSTR + kb];
        }
        BFrag b;
        b.q[0] = *(const U4*)&bptr[kk];
        b.q[1] = *(const U4*)&bptr[kk + 16];
        acc = __builtin_amdgcn_wmma_f32_16x16x32_bf16(
            false, a.v, false, b.v, (short)0, acc, false, false);
      }
      y_acc[tt] = acc;
    }
    __syncthreads();
  }

  // weight rows by routing weight and scatter to partial buffer yp[k]
#pragma unroll
  for (int tt = 0; tt < TPW; ++tt) {
    const int j = (wave * TPW + tt) * 16;
#pragma unroll
    for (int v = 0; v < 8; ++v) {
      int row = v + 8 * s;
      int tk = tokL[row];
      if (tk >= 0) {
        yp[((long long)kL[row] * NTOK + tk) * CDIM + j + m16] =
            wL[row] * y_acc[tt][v];
      }
    }
  }
}

// ---------------------------------------------------------------------------
// 5) Combine the two top-k partials:  out = yp0 + yp1
// ---------------------------------------------------------------------------
__global__ __launch_bounds__(256) void moe_combine_kernel(
    const float* __restrict__ yp, float* __restrict__ out) {
  int i = blockIdx.x * blockDim.x + threadIdx.x;
  int stride = gridDim.x * blockDim.x;
  const int total = NTOK * CDIM;
  for (int o = i; o < total; o += stride)
    out[o] = yp[o] + yp[total + o];
}

// ---------------------------------------------------------------------------
// 6) Aux loss finalize: fixed-order deterministic sum of block partials
// ---------------------------------------------------------------------------
__global__ void moe_aux_kernel(const float* __restrict__ blocksums,
                               float* __restrict__ out_aux) {
  float a = 0.f;
  for (int e = 0; e < NEXP; ++e) {
    float ps = 0.f;
    for (int b = 0; b < GBLKS; ++b) ps += blocksums[b * NEXP + e];
    float f = ps / (float)NTOK;
    a += f * f;
  }
  *out_aux = 0.01f * (float)NEXP * a;
}

// ---------------------------------------------------------------------------
extern "C" void kernel_launch(void* const* d_in, const int* in_sizes, int n_in,
                              void* d_out, int out_size, void* d_ws, size_t ws_size,
                              hipStream_t stream) {
  const float* x  = (const float*)d_in[0];
  const float* gw = (const float*)d_in[1];
  const float* w1 = (const float*)d_in[2];
  const float* w2 = (const float*)d_in[3];
  float* out = (float*)d_out;

  char* ws = (char*)d_ws;
  size_t off = 0;
  unsigned short* xb  = (unsigned short*)(ws + off); off += (size_t)NTOK * CDIM * 2;
  unsigned short* w1t = (unsigned short*)(ws + off); off += (size_t)NEXP * CDIM * FDIM * 2;
  unsigned short* w2t = (unsigned short*)(ws + off); off += (size_t)NEXP * FDIM * CDIM * 2;
  float* blocksums = (float*)(ws + off); off += (size_t)GBLKS * NEXP * 4;
  int*   cnt       = (int*)(ws + off);   off += 256;   // 8 ints, padded
  int*   tileoff   = (int*)(ws + off);   off += 256;   // 9 ints, padded
  int*   btok      = (int*)(ws + off);   off += (size_t)NEXP * CAP * 4;
  float* bw        = (float*)(ws + off); off += (size_t)NEXP * CAP * 4;
  float* yp        = (float*)(ws + off); off += (size_t)2 * NTOK * CDIM * 4;

  moe_convert_kernel<<<4096, 256, 0, stream>>>(x, w1, w2, xb, w1t, w2t, cnt);
  moe_gate_kernel<<<GBLKS, 256, 0, stream>>>(x, gw, cnt, btok, bw, blocksums);
  moe_tileoff_kernel<<<1, 1, 0, stream>>>(cnt, tileoff);
  moe_mlp_kernel<<<MAXTILES, 256, 0, stream>>>(xb, w1t, w2t, btok, bw, cnt,
                                               tileoff, yp);
  moe_combine_kernel<<<1024, 256, 0, stream>>>(yp, out);
  moe_aux_kernel<<<1, 1, 0, stream>>>(blocksums, out + (size_t)NTOK * CDIM);
}
